// ParticleNet_21139829031582
// MI455X (gfx1250) — matrix operand order repaired and checked by reading, not compiled, not run on hardware
//
#include <hip/hip_runtime.h>
#include <hip/hip_bf16.h>

// ---------------------------------------------------------------------------
// ParticleNet forward for gfx1250 (MI455X, wave32, WMMA).
//
// GEMM: one block = 8 waves = 128 rows x 64 cols of output (wave = 16 rows,
// 4 col tiles). A panel (128x32) staged to LDS in WMMA A-fragment order as
// f16 with fused BN+ReLU / edge-gather, double-buffered so staging of step
// i+1 overlaps the WMMAs of step i (one barrier per step). 64-col weight
// slice staged once per block in B-fragment order (weights are L2-resident;
// activations are the HBM traffic -> 64-wide blocks cut A re-reads 4x vs
// 16-wide). Inner loop per wave: 1 A frag + 4 B frags (ds_load_b128) and
// 4x v_wmma_f32_16x16x32_f16. BN batch stats accumulate atomically in the
// epilogue; bn_finalize folds them into scale/shift applied by the next
// GEMM's staging. All hot-path addressing is 32-bit.
// ---------------------------------------------------------------------------

#define NPTS   65536      // B*P
#define NEDGE  262144     // N*K
#define PPG    128        // points per graph
#define NGRAPH 512
#define KNN    4
#define EPSBN  1e-5f
#define ASTRIDE (128 * 48)   // one A panel: 128 rows, 32 k, row stride 48 halves

typedef __attribute__((ext_vector_type(16))) _Float16 v16h;
typedef __attribute__((ext_vector_type(4)))  _Float16 v4h;
typedef __attribute__((ext_vector_type(8)))  float    v8f;

// Fragment K-permutation for 16-bit A (16x32): lane half 0 holds K {0-7,16-23},
// half 1 holds K {8-15,24-31}; contiguous inside each 8-wide segment, so a
// k-aligned quad stays contiguous after permutation.
__device__ __forceinline__ int aperm(int k) {
    return (k < 8) ? k : (k < 16) ? k + 8 : (k < 24) ? k - 8 : k;
}

// Stage one 128x32 A panel (rows rowBase.., k = k0..k0+31) into LDS as f16 in
// plain k order, row stride 48. Quad-vectorized: 1024 float4, 4 per thread.
template<int AMODE, int ACT>
__device__ __forceinline__ void stage_a(_Float16* __restrict__ dstA,
                                        const float* __restrict__ A,
                                        const int*   __restrict__ srcIdx,
                                        const float* __restrict__ scaleIn,
                                        const float* __restrict__ shiftIn,
                                        int rowBase, int k0, int Kd, int Cf, int tid)
{
#pragma unroll
    for (int i = 0; i < 4; ++i) {
        const int idx = i * 256 + tid;
        const int r   = idx >> 3;            // 0..127
        const int kq  = (idx & 7) << 2;      // 0,4,..,28
        const int gk  = k0 + kq;
        float4 va = make_float4(0.f, 0.f, 0.f, 0.f);
        if (AMODE == 0) {
            if (gk < Kd) {                   // zero-pad when Kd not mult of 32 (Kd=16)
                va = *(const float4*)&A[(unsigned)(rowBase + r) * (unsigned)Kd + gk];
                if (ACT == 2) {
                    va.x = va.x * scaleIn[gk+0] + shiftIn[gk+0]; va.x = va.x > 0.f ? va.x : 0.f;
                    va.y = va.y * scaleIn[gk+1] + shiftIn[gk+1]; va.y = va.y > 0.f ? va.y : 0.f;
                    va.z = va.z * scaleIn[gk+2] + shiftIn[gk+2]; va.z = va.z > 0.f ? va.z : 0.f;
                    va.w = va.w * scaleIn[gk+3] + shiftIn[gk+3]; va.w = va.w > 0.f ? va.w : 0.f;
                }
            }
        } else {
            // edge rows: [xi | xj - xi]; Cf multiple of 4 -> quad never straddles
            const int e = rowBase + r;
            const unsigned t = (unsigned)(e >> 2);          // tgt node = e/4
            if (gk < Cf) {
                va = *(const float4*)&A[t * (unsigned)Cf + gk];
            } else {
                const unsigned s = (unsigned)srcIdx[e];
                const int c = gk - Cf;
                const float4 xj = *(const float4*)&A[s * (unsigned)Cf + c];
                const float4 xi = *(const float4*)&A[t * (unsigned)Cf + c];
                va = make_float4(xj.x - xi.x, xj.y - xi.y, xj.z - xi.z, xj.w - xi.w);
            }
        }
        v4h h;
        h[0] = (_Float16)va.x; h[1] = (_Float16)va.y;
        h[2] = (_Float16)va.z; h[3] = (_Float16)va.w;
        *(v4h*)&dstA[r * 48 + kq] = h;               // 8B-aligned ds_store_b64
    }
}

// ---------------------------------------------------------------------------
// AMODE 0: A[row][k] (plain), ACT 0 none / 2 fused affine+relu (prev layer BN)
// AMODE 1: edge rows via srcIdx (tgt(e)=e>>2, K=4 consecutive edges per node)
// Grid: (M/128, Nc/64), 256 threads.
// ---------------------------------------------------------------------------
template<int AMODE, int ACT>
__global__ __launch_bounds__(256)
void gemm_wmma(const float* __restrict__ A,
               const int*   __restrict__ srcIdx,
               const float* __restrict__ W,
               const float* __restrict__ bias,
               const float* __restrict__ scaleIn,
               const float* __restrict__ shiftIn,
               int Kd, int Nc, int Cf,
               float* __restrict__ Hout,
               float* __restrict__ stats,
               int reluStore)
{
    __shared__ _Float16 shA[2 * ASTRIDE];       // double-buffered A panels (24KB)
    __shared__ _Float16 shB[64 * (256 + 16)];   // 64 cols x Kc, stride Kc+16 (~34KB)

    const int tid     = threadIdx.x;
    const int rowBase = blockIdx.x * 128;
    const int colBase = blockIdx.y * 64;
    const int Kc      = (Kd + 31) & ~31;        // K loop rounded to WMMA depth
    const int Ks      = Kc + 16;

    // ---- stage 64-col weight slice as f16, B-fragment order (coalesced on n)
    for (int idx = tid; idx < (Kc << 6); idx += 256) {
        const int k = idx >> 6;
        const int n = idx & 63;
        shB[n * Ks + k] = (k < Kd) ? (_Float16)W[(unsigned)k * (unsigned)Nc + colBase + n]
                                   : (_Float16)0.f;
    }

    const int lane = tid & 31;
    const int wave = tid >> 5;
    const int half = lane >> 4;
    const int lrow = lane & 15;

    stage_a<AMODE, ACT>(shA, A, srcIdx, scaleIn, shiftIn, rowBase, 0, Kd, Cf, tid);
    __syncthreads();

    v8f acc[4] = {{}, {}, {}, {}};
    int cur = 0;
    for (int k0 = 0; k0 < Kc; k0 += 32) {
        if (k0 + 32 < Kc)                        // prefetch next panel into alt buffer
            stage_a<AMODE, ACT>(shA + (cur ^ 1) * ASTRIDE, A, srcIdx, scaleIn, shiftIn,
                                rowBase, k0 + 32, Kd, Cf, tid);
        const v16h a = *(const v16h*)&shA[cur * ASTRIDE + (wave * 16 + lrow) * 48 + half * 16];
#pragma unroll
        for (int j = 0; j < 4; ++j) {
            const v16h b = *(const v16h*)&shB[(j * 16 + lrow) * Ks + k0 + half * 16];
            acc[j] = __builtin_amdgcn_wmma_f32_16x16x32_f16(false, a, false, b,
                                                            (short)0, acc[j], false, false);
        }
        __syncthreads();
        cur ^= 1;
    }

    // ---- epilogue: bias, optional relu, store, BN statistics
#pragma unroll
    for (int j = 0; j < 4; ++j) {
        const int   col = colBase + j * 16 + lrow;
        const float bv  = (bias != nullptr) ? bias[col] : 0.f;
        float s1 = 0.f, s2 = 0.f;
#pragma unroll
        for (int r = 0; r < 8; ++r) {
            const int m = rowBase + wave * 16 + half * 8 + r;   // C/D: row = 8*half + vgpr
            float v = acc[j][r] + bv;
            if (reluStore) v = v > 0.f ? v : 0.f;
            Hout[(unsigned)m * (unsigned)Nc + col] = v;
            s1 += v; s2 += v * v;
        }
        if (stats != nullptr) {
            atomicAdd(&stats[col],      s1);
            atomicAdd(&stats[Nc + col], s2);
        }
    }
}

// ---------------------------------------------------------------------------
__global__ void bn_finalize(const float* __restrict__ stats,
                            const float* __restrict__ gamma,
                            const float* __restrict__ beta,
                            float invM, int C,
                            float* __restrict__ scale, float* __restrict__ shift)
{
    const int c = blockIdx.x * blockDim.x + threadIdx.x;
    if (c >= C) return;
    const float mean = stats[c] * invM;
    const float var  = stats[C + c] * invM - mean * mean;
    const float inv  = gamma[c] * rsqrtf(var + EPSBN);
    scale[c] = inv;
    shift[c] = beta[c] - mean * inv;
}

__global__ void zero_kernel(float* __restrict__ p, int n)
{
    const int i = blockIdx.x * blockDim.x + threadIdx.x;
    if (i < n) p[i] = 0.f;
}

// ---------------------------------------------------------------------------
// GraphNorm: per (graph, channel) mean over 128 points; out = x - ms*mean;
// var over points; y = w*out*rsqrt(var+eps) + b.
// ---------------------------------------------------------------------------
__global__ void graph_norm(const float* __restrict__ X,
                           const float* __restrict__ w,
                           const float* __restrict__ b,
                           const float* __restrict__ ms,
                           int C, float* __restrict__ Y)
{
    const int g = blockIdx.x;
    const int c = threadIdx.x;
    if (c >= C) return;
    const float* xg = X + (unsigned)g * PPG * C;
    float sum = 0.f;
    for (int p = 0; p < PPG; ++p) sum += xg[p * C + c];
    const float m2 = ms[c] * (sum * (1.f / PPG));
    float var = 0.f;
    for (int p = 0; p < PPG; ++p) { const float o = xg[p * C + c] - m2; var += o * o; }
    var *= (1.f / PPG);
    const float inv = w[c] * rsqrtf(var + EPSBN);
    const float sh  = b[c];
    float* yg = Y + (unsigned)g * PPG * C;
    for (int p = 0; p < PPG; ++p) { const float o = xg[p * C + c] - m2; yg[p * C + c] = o * inv + sh; }
}

// ---------------------------------------------------------------------------
// kNN (k=4) within each graph; points staged in LDS; float4 distance loop.
// ---------------------------------------------------------------------------
__global__ void knn_kernel(const float* __restrict__ X, int C, int* __restrict__ srcIdx)
{
    extern __shared__ float sh[];
    const int g = blockIdx.x;
    const int p = threadIdx.x;                     // 128 threads
    for (int c = p; c < PPG * C; c += PPG) sh[c] = X[(unsigned)g * PPG * C + c];
    __syncthreads();

    float bd[KNN]; int bi[KNN];
#pragma unroll
    for (int k = 0; k < KNN; ++k) { bd[k] = 3.0e38f; bi[k] = 0; }

    for (int q = 0; q < PPG; ++q) {
        float d = 0.f;
        for (int c = 0; c < C; c += 4) {
            const float4 a = *(const float4*)&sh[p * C + c];
            const float4 b = *(const float4*)&sh[q * C + c];
            const float d0 = a.x - b.x, d1 = a.y - b.y, d2 = a.z - b.z, d3 = a.w - b.w;
            d += d0 * d0 + d1 * d1 + d2 * d2 + d3 * d3;
        }
        if (q == p) d += 1.0e9f;                   // mirror reference self-mask
        if (d < bd[KNN - 1]) {                     // strict <: ties keep earlier q
            int j = KNN - 1;
            while (j > 0 && d < bd[j - 1]) { bd[j] = bd[j - 1]; bi[j] = bi[j - 1]; --j; }
            bd[j] = d; bi[j] = q;
        }
    }
#pragma unroll
    for (int k = 0; k < KNN; ++k)
        srcIdx[((unsigned)g * PPG + p) * KNN + k] = g * PPG + bi[k];
}

// ---------------------------------------------------------------------------
// Edge-conv epilogue: out[n] = relu( mean_k relu(bn(Hmlp[4n+k])) + bn(Hsc[n]) )
// ---------------------------------------------------------------------------
__global__ void edge_aggregate(const float* __restrict__ Hm,
                               const float* __restrict__ pS, const float* __restrict__ pSh,
                               const float* __restrict__ Hs,
                               const float* __restrict__ sS, const float* __restrict__ sSh,
                               int C, float* __restrict__ Y)
{
    const int n = blockIdx.x;
    const int c = threadIdx.x;
    if (c >= C) return;
    float s = 0.f;
#pragma unroll
    for (int k = 0; k < KNN; ++k) {
        float h = Hm[((unsigned)n * KNN + k) * (unsigned)C + c];
        h = h * pS[c] + pSh[c];
        s += (h > 0.f ? h : 0.f);
    }
    s *= 0.25f;
    const float sc = Hs[(unsigned)n * C + c] * sS[c] + sSh[c];
    const float v  = s + sc;
    Y[(unsigned)n * C + c] = v > 0.f ? v : 0.f;
}

__global__ void pool_kernel(const float* __restrict__ X, int C, float* __restrict__ G)
{
    const int bb = blockIdx.x;
    const int c  = threadIdx.x;
    if (c >= C) return;
    float s = 0.f;
    for (int p = 0; p < PPG; ++p) s += X[((unsigned)bb * PPG + p) * C + c];
    G[(unsigned)bb * C + c] = s * (1.f / PPG);
}

__global__ void out_softmax(const float* __restrict__ D2,
                            const float* __restrict__ Wo, const float* __restrict__ bo,
                            float* __restrict__ out)
{
    const int i = blockIdx.x * blockDim.x + threadIdx.x;
    if (i >= NGRAPH) return;
    float l0 = bo[0], l1 = bo[1];
    for (int c = 0; c < 64; ++c) {
        const float v = D2[(unsigned)i * 64 + c];
        l0 += v * Wo[c * 2 + 0];
        l1 += v * Wo[c * 2 + 1];
    }
    const float mx = l0 > l1 ? l0 : l1;
    const float e0 = __expf(l0 - mx), e1 = __expf(l1 - mx);
    const float inv = 1.f / (e0 + e1);
    out[i * 2 + 0] = e0 * inv;
    out[i * 2 + 1] = e1 * inv;
}

__global__ void cvt_idx(const long long* __restrict__ ei, int E, int* __restrict__ srcIdx)
{
    const int e = blockIdx.x * blockDim.x + threadIdx.x;
    if (e < E) srcIdx[e] = (int)ei[e];
}

// ---------------------------------------------------------------------------
extern "C" void kernel_launch(void* const* d_in, const int* in_sizes, int n_in,
                              void* d_out, int out_size, void* d_ws, size_t ws_size,
                              hipStream_t stream)
{
    (void)in_sizes; (void)n_in; (void)out_size; (void)ws_size;

    // inputs: x, edge_index, batch, then params flattened pytree-style (dict
    // keys sorted): conv1(mlp l0..l2 (W,b,g,be), sc(Ws,bs,gs,bes)), conv2,
    // conv3, dense1(W,b), dense2(W,b), gn0(b,ms,w), gn1, gn2, gn3, output(W,b).
    auto F = [&](int i) { return (const float*)d_in[i]; };
    const float*     x  = F(0);
    const long long* ei = (const long long*)d_in[1];

    char* wp = (char*)d_ws;
    auto alloc = [&](size_t bytes) {
        void* p = (void*)wp;
        wp += (bytes + 255) & ~(size_t)255;
        return p;
    };
    float* nbuf0 = (float*)alloc((size_t)NPTS  * 128 * 4);
    float* nbuf1 = (float*)alloc((size_t)NPTS  * 128 * 4);
    float* Hsc   = (float*)alloc((size_t)NPTS  * 128 * 4);
    float* H0    = (float*)alloc((size_t)NEDGE * 128 * 4);
    float* H1    = (float*)alloc((size_t)NEDGE * 128 * 4);
    int*   sidx  = (int*)  alloc((size_t)NEDGE * 4);
    float* stats = (float*)alloc(2 * 128 * 4);
    float* bnPs  = (float*)alloc(128 * 4);
    float* bnPh  = (float*)alloc(128 * 4);
    float* bnSs  = (float*)alloc(128 * 4);
    float* bnSh  = (float*)alloc(128 * 4);
    float* G     = (float*)alloc((size_t)NGRAPH * 128 * 4);
    float* D1    = (float*)alloc((size_t)NGRAPH * 64 * 4);
    float* D2    = (float*)alloc((size_t)NGRAPH * 64 * 4);

    auto gemm = [&](int amode, int act,
                    const float* A, const int* si, const float* W, const float* bias,
                    const float* scIn, const float* shIn,
                    int M, int Kd, int Nc, int Cf,
                    float* Hout, float* st, int reluStore) {
        if (st) zero_kernel<<<1, 256, 0, stream>>>(st, 2 * Nc);
        dim3 grid(M / 128, Nc / 64);
        if (amode == 1)
            gemm_wmma<1, 0><<<grid, 256, 0, stream>>>(A, si, W, bias, scIn, shIn,
                                                      Kd, Nc, Cf, Hout, st, reluStore);
        else if (act == 2)
            gemm_wmma<0, 2><<<grid, 256, 0, stream>>>(A, si, W, bias, scIn, shIn,
                                                      Kd, Nc, Cf, Hout, st, reluStore);
        else
            gemm_wmma<0, 0><<<grid, 256, 0, stream>>>(A, si, W, bias, scIn, shIn,
                                                      Kd, Nc, Cf, Hout, st, reluStore);
    };

    auto run_conv = [&](const float* Xn, int Cin, int Cout, int pb, float* Yout) {
        const float *W0 = F(pb+0),  *b0 = F(pb+1),  *g0 = F(pb+2),  *e0 = F(pb+3);
        const float *W1 = F(pb+4),  *b1 = F(pb+5),  *g1 = F(pb+6),  *e1 = F(pb+7);
        const float *W2 = F(pb+8),  *b2 = F(pb+9),  *g2 = F(pb+10), *e2 = F(pb+11);
        const float *Ws = F(pb+12), *bs = F(pb+13), *gs = F(pb+14), *es = F(pb+15);

        // shortcut projection: Hsc = Xn @ Ws + bs  (stats over N rows)
        gemm(0, 0, Xn, nullptr, Ws, bs, nullptr, nullptr, NPTS, Cin, Cout, 0, Hsc, stats, 0);
        bn_finalize<<<1, 128, 0, stream>>>(stats, gs, es, 1.f / NPTS, Cout, bnSs, bnSh);

        // mlp layer 0: edge-gather A, K = 2*Cin
        gemm(1, 0, Xn, sidx, W0, b0, nullptr, nullptr, NEDGE, 2 * Cin, Cout, Cin, H0, stats, 0);
        bn_finalize<<<1, 128, 0, stream>>>(stats, g0, e0, 1.f / NEDGE, Cout, bnPs, bnPh);

        // mlp layer 1/2: fused bn+relu on input
        gemm(0, 2, H0, nullptr, W1, b1, bnPs, bnPh, NEDGE, Cout, Cout, 0, H1, stats, 0);
        bn_finalize<<<1, 128, 0, stream>>>(stats, g1, e1, 1.f / NEDGE, Cout, bnPs, bnPh);

        gemm(0, 2, H1, nullptr, W2, b2, bnPs, bnPh, NEDGE, Cout, Cout, 0, H0, stats, 0);
        bn_finalize<<<1, 128, 0, stream>>>(stats, g2, e2, 1.f / NEDGE, Cout, bnPs, bnPh);

        edge_aggregate<<<NPTS, Cout, 0, stream>>>(H0, bnPs, bnPh, Hsc, bnSs, bnSh, Cout, Yout);
    };

    // ---- forward pipeline -------------------------------------------------
    graph_norm<<<NGRAPH, 32, 0, stream>>>(x, F(57), F(55), F(56), 16, nbuf0);   // gn0

    cvt_idx<<<(NEDGE + 255) / 256, 256, 0, stream>>>(ei, NEDGE, sidx);
    run_conv(nbuf0, 16, 64, 3, nbuf1);                                          // conv1

    graph_norm<<<NGRAPH, 64, 0, stream>>>(nbuf1, F(60), F(58), F(59), 64, nbuf0); // gn1

    knn_kernel<<<NGRAPH, PPG, PPG * 64 * 4, stream>>>(nbuf0, 64, sidx);
    run_conv(nbuf0, 64, 128, 19, nbuf1);                                        // conv2

    graph_norm<<<NGRAPH, 128, 0, stream>>>(nbuf1, F(63), F(61), F(62), 128, nbuf0); // gn2

    knn_kernel<<<NGRAPH, PPG, PPG * 128 * 4, stream>>>(nbuf0, 128, sidx);
    run_conv(nbuf0, 128, 128, 35, nbuf1);                                       // conv3

    graph_norm<<<NGRAPH, 128, 0, stream>>>(nbuf1, F(66), F(64), F(65), 128, nbuf0); // gn3

    pool_kernel<<<NGRAPH, 128, 0, stream>>>(nbuf0, 128, G);
    gemm(0, 0, G,  nullptr, F(51), F(52), nullptr, nullptr, NGRAPH, 128, 64, 0, D1, nullptr, 1);
    gemm(0, 0, D1, nullptr, F(53), F(54), nullptr, nullptr, NGRAPH, 64,  64, 0, D2, nullptr, 1);
    out_softmax<<<(NGRAPH + 255) / 256, 256, 0, stream>>>(D2, F(67), F(68), (float*)d_out);
}
